// spatMN_52329881535021
// MI455X (gfx1250) — matrix operand order repaired and checked by reading, not compiled, not run on hardware
//
#include <hip/hip_runtime.h>
#include <hip/hip_bf16.h>

// ---------------------------------------------------------------------------
// BF16 WMMA GEMM engine for gfx1250 (wave32, V_WMMA_F32_16X16X32_BF16).
// Workgroup = 256 threads = 8 waves. Tile: 256(M) x 32(N); each wave owns
// 32(M) x 32(N) (2 A-frags x 2 B-frags = 4 WMMAs per 32-wide K panel).
//
// Pipeline per K panel (double-buffered LDS):
//   1) ds_load_b128 x8: fragments from current buffer   (issued first)
//   2) global gathers of next panel into registers      (latency off critical path)
//   3) 4x v_wmma (waits only on step-1 ds loads)
//   4) v_cvt_pk_bf16_f32 + ds_store_b32 next panel into other buffer
//      (overlaps the matrix pipe)
//   5) s_barrier
// LDS tiles are in WMMA fragment order: 48B slot per (subtile,lane) holding the
// lane's 16 bf16 fragment elements contiguously -> frag load = 2x ds_load_b128.
// ---------------------------------------------------------------------------

typedef __attribute__((ext_vector_type(16))) __bf16        v16bf;
typedef __attribute__((ext_vector_type(2)))  __bf16        v2bf;
typedef __attribute__((ext_vector_type(8)))  float         v8f;
typedef __attribute__((ext_vector_type(4)))  unsigned int  v4u;

#define ASLOT 24            // ushorts per lane slot: 32 data bytes + 16 pad
#define ABUF  (16 * 32 * ASLOT)
#define BBUF  ( 2 * 32 * ASLOT)

// Packed f32x2 -> bf16x2: exact pattern for v_cvt_pk_bf16_f32 (no shift/or merge).
static __device__ __forceinline__ unsigned pack2(float a, float b) {
  v2bf p;
  p[0] = (__bf16)a;
  p[1] = (__bf16)b;
  return __builtin_bit_cast(unsigned, p);
}

union FragLd { v16bf v; v4u q[2]; };
static __device__ __forceinline__ v16bf ld_frag(const unsigned short* p) {
  FragLd f;
  f.q[0] = *(const v4u*)(p);        // ds_load_b128
  f.q[1] = *(const v4u*)(p + 8);    // ds_load_b128
  return f.v;
}

#define WMMA_BF16(A, B, C) \
  __builtin_amdgcn_wmma_f32_16x16x32_bf16(false, (A), false, (B), (short)0, (C), false, false)

// Fragment-order slot math (ISA 7.12.2). For a fixed K column kc the A-side
// (group, element) are per-thread constants.
static __device__ __forceinline__ int a_g16(int kc) { return ((kc < 16) ? (kc >> 3) : ((kc - 16) >> 3)) * 16; }
static __device__ __forceinline__ int a_e(int kc)   { return (kc < 16) ? (kc & 7) : (8 + (kc & 7)); }
// ky from r=k%9 without division: two compares.
static __device__ __forceinline__ int ky_of(int r)  { return (r >= 6) ? 2 : (r >= 3 ? 1 : 0); }

// ---------------------------------------------------------------------------
// Kernel 1: implicit-GEMM 3x3 conv (+bias, +ReLU). Cout = 32.
//   M = B*H*H (grid.x tiles of 256), N = 32, K = Cin*9 zero-padded to Kpad.
// ---------------------------------------------------------------------------
__global__ __launch_bounds__(256) void conv3x3_wmma(
    const float* __restrict__ x, const float* __restrict__ w,
    const float* __restrict__ bias, float* __restrict__ out,
    int Cin, int H, int pad, int dil, int Kdim, int Kpad)
{
  const int HW = H * H;
  const int tileM = blockIdx.x;
  __shared__ __align__(16) unsigned short ldsA[2][ABUF];
  __shared__ __align__(16) unsigned short ldsB[2][BBUF];
  const int tid = threadIdx.x, lane = tid & 31, wave = tid >> 5;

  const int kc = (tid & 15) * 2;               // fixed K column pair per panel
  const int ge = a_g16(kc), ea = a_e(kc);

  int rowbase[16], oyox[16], woff[16], boff[2];
#pragma unroll
  for (int i = 0; i < 16; ++i) {
    const int mr = (tid >> 4) + 16 * i;
    const int m = tileM * 256 + mr;
    const int b = m / HW, s = m - b * HW;
    const int oy = s / H, ox = s - oy * H;
    rowbase[i] = (b * Cin) * HW + oy * H + ox;
    oyox[i] = (oy << 16) | ox;
    woff[i] = ((mr >> 4) * 32 + ge + (mr & 15)) * ASLOT + ea;
  }
#pragma unroll
  for (int it = 0; it < 2; ++it) {
    const int nc = (tid >> 4) + it * 16;        // B panel: kr == kc for this thread
    boff[it] = ((nc >> 4) * 32 + ge + (nc & 15)) * ASLOT + ea;
  }

  float ga[32], gbv[4];
  auto gatherA = [&](int k0) {
    const int k = k0 + kc;
    const bool kv = k < Kdim;                   // Kdim even, kc even -> covers k+1
    const int c0 = k / 9, r0 = k - c0 * 9;
    int r1 = r0 + 1, c1 = c0; if (r1 == 9) { r1 = 0; ++c1; }
    const int ky0 = ky_of(r0), kx0 = r0 - 3 * ky0;
    const int ky1 = ky_of(r1), kx1 = r1 - 3 * ky1;
    const int dy0 = ky0 * dil - pad, dx0 = kx0 * dil - pad;
    const int dy1 = ky1 * dil - pad, dx1 = kx1 * dil - pad;
    const int aoff0 = c0 * HW + dy0 * H + dx0;  // thread-uniform per panel
    const int aoff1 = c1 * HW + dy1 * H + dx1;
#pragma unroll
    for (int i = 0; i < 16; ++i) {
      const int oy = oyox[i] >> 16, ox = oyox[i] & 0xffff;
      float v0 = 0.f, v1 = 0.f;
      if (kv && (unsigned)(oy + dy0) < (unsigned)H && (unsigned)(ox + dx0) < (unsigned)H)
        v0 = x[rowbase[i] + aoff0];
      if (kv && (unsigned)(oy + dy1) < (unsigned)H && (unsigned)(ox + dx1) < (unsigned)H)
        v1 = x[rowbase[i] + aoff1];
      ga[2 * i] = v0; ga[2 * i + 1] = v1;
    }
  };
  auto gatherB = [&](int k0) {
    const int kk = k0 + kc;
#pragma unroll
    for (int it = 0; it < 2; ++it) {
      const int nc = (tid >> 4) + it * 16;
      float v0 = 0.f, v1 = 0.f;
      if (kk < Kdim) { v0 = w[nc * Kdim + kk]; v1 = w[nc * Kdim + kk + 1]; }
      gbv[2 * it] = v0; gbv[2 * it + 1] = v1;
    }
  };
  auto commit = [&](int buf) {
#pragma unroll
    for (int i = 0; i < 16; ++i) *(unsigned*)&ldsA[buf][woff[i]] = pack2(ga[2 * i], ga[2 * i + 1]);
#pragma unroll
    for (int it = 0; it < 2; ++it) *(unsigned*)&ldsB[buf][boff[it]] = pack2(gbv[2 * it], gbv[2 * it + 1]);
  };

  v8f acc00 = {}, acc01 = {}, acc10 = {}, acc11 = {};

  gatherA(0); gatherB(0); commit(0);
  __syncthreads();
  for (int k0 = 0; k0 < Kpad; k0 += 32) {
    const int cur = (k0 >> 5) & 1;
    const unsigned short* aw = ldsA[cur] + (wave * 2) * 32 * ASLOT;
    const v16bf a0 = ld_frag(aw + lane * ASLOT);
    const v16bf a1 = ld_frag(aw + (32 + lane) * ASLOT);
    const v16bf b0 = ld_frag(ldsB[cur] + lane * ASLOT);
    const v16bf b1 = ld_frag(ldsB[cur] + (32 + lane) * ASLOT);
    const bool more = (k0 + 32) < Kpad;
    if (more) { gatherA(k0 + 32); gatherB(k0 + 32); }
    acc00 = WMMA_BF16(a0, b0, acc00);
    acc01 = WMMA_BF16(a0, b1, acc01);
    acc10 = WMMA_BF16(a1, b0, acc10);
    acc11 = WMMA_BF16(a1, b1, acc11);
    if (more) commit(cur ^ 1);
    __syncthreads();
  }

  // epilogue -> (B, 32, H, H), bias + ReLU
  const int nb = lane & 15, g = lane >> 4;
  const int m0 = tileM * 256 + wave * 32;
#pragma unroll
  for (int v = 0; v < 8; ++v) {
    int m = m0 + v + 8 * g;
    int b = m / HW, s = m - b * HW;
    float r;
    r = acc00[v] + bias[nb];      out[(b * 32 + nb)      * HW + s] = r > 0.f ? r : 0.f;
    r = acc01[v] + bias[16 + nb]; out[(b * 32 + 16 + nb) * HW + s] = r > 0.f ? r : 0.f;
    m = m0 + 16 + v + 8 * g;
    b = m / HW; s = m - b * HW;
    r = acc10[v] + bias[nb];      out[(b * 32 + nb)      * HW + s] = r > 0.f ? r : 0.f;
    r = acc11[v] + bias[16 + nb]; out[(b * 32 + 16 + nb) * HW + s] = r > 0.f ? r : 0.f;
  }
}

// ---------------------------------------------------------------------------
// Kernel 2: ConvLSTM gate GEMM, one timestep = patch (pj,pk) of the 27x27 map.
//   M = B*81 (grid.x: 81 tiles of 256), N = 128 (grid.y: 4), K = 576.
//   K panels [0,288): x-source; [288,576): h-source (uniform per panel).
// ---------------------------------------------------------------------------
__global__ __launch_bounds__(256) void lstm_gates_wmma(
    const float* __restrict__ xfull, const float* __restrict__ hbuf,
    const float* __restrict__ lw, const float* __restrict__ lb,
    float* __restrict__ gates, int pj, int pk)
{
  const int tileM = blockIdx.x;
  const int nbase = blockIdx.y * 32;
  __shared__ __align__(16) unsigned short ldsA[2][ABUF];
  __shared__ __align__(16) unsigned short ldsB[2][BBUF];
  const int tid = threadIdx.x, lane = tid & 31, wave = tid >> 5;

  const int kc = (tid & 15) * 2;
  const int ge = a_g16(kc), ea = a_e(kc);

  int xbase[16], hbase[16], oyox[16], woff[16], boff[2];
#pragma unroll
  for (int i = 0; i < 16; ++i) {
    const int mr = (tid >> 4) + 16 * i;
    const int m = tileM * 256 + mr;
    const int b = m / 81, s = m - b * 81;
    const int oy = s / 9, ox = s - oy * 9;
    xbase[i] = b * 23328 + (pj * 9 + oy - 1) * 27 + (pk * 9 + ox - 1);  // (B,32,27,27)
    hbase[i] = b * 2592 + (oy - 1) * 9 + (ox - 1);                      // (B,32,9,9)
    oyox[i] = (oy << 16) | ox;
    woff[i] = ((mr >> 4) * 32 + ge + (mr & 15)) * ASLOT + ea;
  }
#pragma unroll
  for (int it = 0; it < 2; ++it) {
    const int nc = (tid >> 4) + it * 16;
    boff[it] = ((nc >> 4) * 32 + ge + (nc & 15)) * ASLOT + ea;
  }

  float ga[32], gbv[4];
  auto gatherA = [&](int k0) {
    const int kl = (k0 < 288 ? k0 : k0 - 288) + kc;   // phase-local k
    const int c0 = kl / 9, r0 = kl - c0 * 9;
    int r1 = r0 + 1, c1 = c0; if (r1 == 9) { r1 = 0; ++c1; }
    const int ky0 = ky_of(r0), kx0 = r0 - 3 * ky0;
    const int ky1 = ky_of(r1), kx1 = r1 - 3 * ky1;
    if (k0 < 288) {   // x phase: rows stride 27, channels stride 729
      const int o0 = c0 * 729 + ky0 * 27 + kx0;
      const int o1 = c1 * 729 + ky1 * 27 + kx1;
#pragma unroll
      for (int i = 0; i < 16; ++i) {
        const int oy = oyox[i] >> 16, ox = oyox[i] & 0xffff;
        float v0 = 0.f, v1 = 0.f;
        if ((unsigned)(oy - 1 + ky0) < 9u && (unsigned)(ox - 1 + kx0) < 9u) v0 = xfull[xbase[i] + o0];
        if ((unsigned)(oy - 1 + ky1) < 9u && (unsigned)(ox - 1 + kx1) < 9u) v1 = xfull[xbase[i] + o1];
        ga[2 * i] = v0; ga[2 * i + 1] = v1;
      }
    } else {          // h phase: rows stride 9, channels stride 81
      const int o0 = c0 * 81 + ky0 * 9 + kx0;
      const int o1 = c1 * 81 + ky1 * 9 + kx1;
#pragma unroll
      for (int i = 0; i < 16; ++i) {
        const int oy = oyox[i] >> 16, ox = oyox[i] & 0xffff;
        float v0 = 0.f, v1 = 0.f;
        if ((unsigned)(oy - 1 + ky0) < 9u && (unsigned)(ox - 1 + kx0) < 9u) v0 = hbuf[hbase[i] + o0];
        if ((unsigned)(oy - 1 + ky1) < 9u && (unsigned)(ox - 1 + kx1) < 9u) v1 = hbuf[hbase[i] + o1];
        ga[2 * i] = v0; ga[2 * i + 1] = v1;
      }
    }
  };
  auto gatherB = [&](int k0) {
    const int kk = k0 + kc;
#pragma unroll
    for (int it = 0; it < 2; ++it) {
      const int nc = (tid >> 4) + it * 16;
      gbv[2 * it]     = lw[(nbase + nc) * 576 + kk];
      gbv[2 * it + 1] = lw[(nbase + nc) * 576 + kk + 1];
    }
  };
  auto commit = [&](int buf) {
#pragma unroll
    for (int i = 0; i < 16; ++i) *(unsigned*)&ldsA[buf][woff[i]] = pack2(ga[2 * i], ga[2 * i + 1]);
#pragma unroll
    for (int it = 0; it < 2; ++it) *(unsigned*)&ldsB[buf][boff[it]] = pack2(gbv[2 * it], gbv[2 * it + 1]);
  };

  v8f acc00 = {}, acc01 = {}, acc10 = {}, acc11 = {};

  gatherA(0); gatherB(0); commit(0);
  __syncthreads();
  for (int k0 = 0; k0 < 576; k0 += 32) {
    const int cur = (k0 >> 5) & 1;
    const unsigned short* aw = ldsA[cur] + (wave * 2) * 32 * ASLOT;
    const v16bf a0 = ld_frag(aw + lane * ASLOT);
    const v16bf a1 = ld_frag(aw + (32 + lane) * ASLOT);
    const v16bf b0 = ld_frag(ldsB[cur] + lane * ASLOT);
    const v16bf b1 = ld_frag(ldsB[cur] + (32 + lane) * ASLOT);
    const bool more = (k0 + 32) < 576;
    if (more) { gatherA(k0 + 32); gatherB(k0 + 32); }
    acc00 = WMMA_BF16(a0, b0, acc00);
    acc01 = WMMA_BF16(a0, b1, acc01);
    acc10 = WMMA_BF16(a1, b0, acc10);
    acc11 = WMMA_BF16(a1, b1, acc11);
    if (more) commit(cur ^ 1);
    __syncthreads();
  }

  const int nb = lane & 15, g = lane >> 4;
  const int m0 = tileM * 256 + wave * 32;
#pragma unroll
  for (int v = 0; v < 8; ++v) {
    int m = m0 + v + 8 * g;
    int b = m / 81, s = m - b * 81;
    gates[(b * 128 + nbase + nb)      * 81 + s] = acc00[v] + lb[nbase + nb];
    gates[(b * 128 + nbase + 16 + nb) * 81 + s] = acc01[v] + lb[nbase + 16 + nb];
    m = m0 + 16 + v + 8 * g;
    b = m / 81; s = m - b * 81;
    gates[(b * 128 + nbase + nb)      * 81 + s] = acc10[v] + lb[nbase + nb];
    gates[(b * 128 + nbase + 16 + nb) * 81 + s] = acc11[v] + lb[nbase + 16 + nb];
  }
}

// ---------------------------------------------------------------------------
// Kernel 3: ConvLSTM elementwise cell update. total = B*32*81.
// ---------------------------------------------------------------------------
static __device__ __forceinline__ float sigmf(float x) { return 1.f / (1.f + __expf(-x)); }

__global__ __launch_bounds__(256) void lstm_update(
    const float* __restrict__ gates, float* __restrict__ c, float* __restrict__ h, int total)
{
  const int idx = blockIdx.x * blockDim.x + threadIdx.x;
  if (idx >= total) return;
  const int s = idx % 81;
  const int t = idx / 81;
  const int ch = t % 32;
  const int b = t / 32;
  const float* gb = gates + (size_t)b * 128 * 81;
  const float gi = gb[(ch)      * 81 + s];
  const float gf = gb[(32 + ch) * 81 + s];
  const float go = gb[(64 + ch) * 81 + s];
  const float gg = gb[(96 + ch) * 81 + s];
  const float cn = sigmf(gf) * c[idx] + sigmf(gi) * tanhf(gg);
  c[idx] = cn;
  h[idx] = sigmf(go) * tanhf(cn);
}

// ---------------------------------------------------------------------------
// Kernel 4: dense FC GEMM  out = [relu]( A(256,K) @ Wt(N,K)^T + bias [+ out] )
//   Single M-tile (M = 256); grid.y tiles N by 32 (bounds-checked for N=16).
// ---------------------------------------------------------------------------
__global__ __launch_bounds__(256) void fc_wmma(
    const float* __restrict__ A, const float* __restrict__ Wt,
    const float* __restrict__ bias, float* __restrict__ out,
    int N, int K, int relu, int accum)
{
  const int nbase = blockIdx.y * 32;
  __shared__ __align__(16) unsigned short ldsA[2][ABUF];
  __shared__ __align__(16) unsigned short ldsB[2][BBUF];
  const int tid = threadIdx.x, lane = tid & 31, wave = tid >> 5;

  const int kc = (tid & 15) * 2;
  const int ge = a_g16(kc), ea = a_e(kc);

  int rowA[16], woff[16], boff[2];
#pragma unroll
  for (int i = 0; i < 16; ++i) {
    const int mr = (tid >> 4) + 16 * i;
    rowA[i] = mr * K;
    woff[i] = ((mr >> 4) * 32 + ge + (mr & 15)) * ASLOT + ea;
  }
#pragma unroll
  for (int it = 0; it < 2; ++it) {
    const int nc = (tid >> 4) + it * 16;
    boff[it] = ((nc >> 4) * 32 + ge + (nc & 15)) * ASLOT + ea;
  }

  float ga[32], gbv[4];
  auto gatherA = [&](int k0) {
    const int k = k0 + kc;                      // K multiple of 32 -> always valid
#pragma unroll
    for (int i = 0; i < 16; ++i) { ga[2 * i] = A[rowA[i] + k]; ga[2 * i + 1] = A[rowA[i] + k + 1]; }
  };
  auto gatherB = [&](int k0) {
    const int kk = k0 + kc;
#pragma unroll
    for (int it = 0; it < 2; ++it) {
      const int n = nbase + (tid >> 4) + it * 16;
      float v0 = 0.f, v1 = 0.f;
      if (n < N) { v0 = Wt[n * K + kk]; v1 = Wt[n * K + kk + 1]; }
      gbv[2 * it] = v0; gbv[2 * it + 1] = v1;
    }
  };
  auto commit = [&](int buf) {
#pragma unroll
    for (int i = 0; i < 16; ++i) *(unsigned*)&ldsA[buf][woff[i]] = pack2(ga[2 * i], ga[2 * i + 1]);
#pragma unroll
    for (int it = 0; it < 2; ++it) *(unsigned*)&ldsB[buf][boff[it]] = pack2(gbv[2 * it], gbv[2 * it + 1]);
  };

  v8f acc00 = {}, acc01 = {}, acc10 = {}, acc11 = {};

  gatherA(0); gatherB(0); commit(0);
  __syncthreads();
  for (int k0 = 0; k0 < K; k0 += 32) {
    const int cur = (k0 >> 5) & 1;
    const unsigned short* aw = ldsA[cur] + (wave * 2) * 32 * ASLOT;
    const v16bf a0 = ld_frag(aw + lane * ASLOT);
    const v16bf a1 = ld_frag(aw + (32 + lane) * ASLOT);
    const v16bf b0 = ld_frag(ldsB[cur] + lane * ASLOT);
    const v16bf b1 = ld_frag(ldsB[cur] + (32 + lane) * ASLOT);
    const bool more = (k0 + 32) < K;
    if (more) { gatherA(k0 + 32); gatherB(k0 + 32); }
    acc00 = WMMA_BF16(a0, b0, acc00);
    acc01 = WMMA_BF16(a0, b1, acc01);
    acc10 = WMMA_BF16(a1, b0, acc10);
    acc11 = WMMA_BF16(a1, b1, acc11);
    if (more) commit(cur ^ 1);
    __syncthreads();
  }

  const int nb = lane & 15, g = lane >> 4;
  const int m0 = wave * 32;
#pragma unroll
  for (int v = 0; v < 8; ++v) {
#pragma unroll
    for (int half = 0; half < 2; ++half) {
      const int m = m0 + 16 * half + v + 8 * g;
      const v8f& aL = half ? acc10 : acc00;
      const v8f& aH = half ? acc11 : acc01;
      const int n0 = nbase + nb, n1 = nbase + 16 + nb;
      if (n0 < N) {
        float r = aL[v] + bias[n0];
        if (accum) r += out[m * N + n0];
        if (relu && r < 0.f) r = 0.f;
        out[m * N + n0] = r;
      }
      if (n1 < N) {
        float r = aH[v] + bias[n1];
        if (accum) r += out[m * N + n1];
        if (relu && r < 0.f) r = 0.f;
        out[m * N + n1] = r;
      }
    }
  }
}

// ---------------------------------------------------------------------------
// Host orchestration (graph-capture safe: kernel launches + hipMemsetAsync only)
// ---------------------------------------------------------------------------
extern "C" void kernel_launch(void* const* d_in, const int* in_sizes, int n_in,
                              void* d_out, int out_size, void* d_ws, size_t ws_size,
                              hipStream_t stream) {
  (void)in_sizes; (void)n_in; (void)out_size; (void)ws_size;
  const float* x_spat = (const float*)d_in[0];
  const float* fc2_w  = (const float*)d_in[25];
  const float* fc2_b  = (const float*)d_in[26];
  const float* pre_w  = (const float*)d_in[27];
  const float* pre_b  = (const float*)d_in[28];

  char* ws = (char*)d_ws;
  size_t off = 0;
  auto alloc = [&](size_t bytes) -> float* {
    float* p = (float*)(ws + off);
    off += (bytes + 255) & ~(size_t)255;
    return p;
  };
  const size_t XB = (size_t)256 * 32 * 729 * 4;   // (B,32,27,27)
  const size_t HB = (size_t)256 * 32 * 81 * 4;    // (B,32,9,9)
  const size_t GB = (size_t)256 * 128 * 81 * 4;   // (B,128,9,9)
  float* xa    = alloc(XB);
  float* xb    = alloc(XB);
  float* hbuf  = alloc(HB);
  float* cbuf  = alloc(HB);
  float* gates = alloc(GB);
  float* ybuf  = alloc((size_t)256 * 128 * 4);
  float* y2    = alloc((size_t)256 * 128 * 4);

  const int MT27 = 186624 / 256;  // 729 M-tiles over B*27*27
  const int MT9  = 20736 / 256;   // 81  M-tiles over B*9*9

  for (int blk = 0; blk < 3; ++blk) {
    const int base = 1 + blk * 8;
    const float* w1 = (const float*)d_in[base + 0];
    const float* b1 = (const float*)d_in[base + 1];
    const float* w2 = (const float*)d_in[base + 2];
    const float* b2 = (const float*)d_in[base + 3];
    const float* lw = (const float*)d_in[base + 4];
    const float* lb = (const float*)d_in[base + 5];
    const float* fw = (const float*)d_in[base + 6];
    const float* fb = (const float*)d_in[base + 7];

    const float* cin_ptr = (blk == 0) ? x_spat : xb;
    const int Cin  = (blk == 0) ? 200 : 32;
    const int Kdim = Cin * 9;
    const int Kpad = (Kdim + 31) & ~31;

    conv3x3_wmma<<<dim3(MT27), 256, 0, stream>>>(cin_ptr, w1, b1, xa, Cin, 27, 1, 1, Kdim, Kpad);
    conv3x3_wmma<<<dim3(MT27), 256, 0, stream>>>(xa,      w2, b2, xb, 32,  27, 2, 2, 288,  288);

    hipMemsetAsync(hbuf, 0, HB, stream);
    hipMemsetAsync(cbuf, 0, HB, stream);
    for (int t = 0; t < 9; ++t) {
      lstm_gates_wmma<<<dim3(MT9, 4), 256, 0, stream>>>(xb, hbuf, lw, lb, gates, t / 3, t % 3);
      const int total = 256 * 32 * 81;
      lstm_update<<<(total + 255) / 256, 256, 0, stream>>>(gates, cbuf, hbuf, total);
    }
    // y (+)= h_flat(B,2592) @ fw^T ; hbuf layout (b, c*81+s) matches reshape order
    fc_wmma<<<dim3(1, 4), 256, 0, stream>>>(hbuf, fw, fb, ybuf, 128, 2592, 0, blk > 0);
  }

  fc_wmma<<<dim3(1, 4), 256, 0, stream>>>(ybuf, fc2_w, fc2_b, y2, 128, 128, 1, 0);
  fc_wmma<<<dim3(1, 1), 256, 0, stream>>>(y2, pre_w, pre_b, (float*)d_out, 16, 128, 0, 0);
}
